// RG_LRU_48962627174697
// MI455X (gfx1250) — compile-verified, matrix operand order checked by reading, not run on hardware
//
#include <hip/hip_runtime.h>

// ---------------------------------------------------------------------------
// RG-LRU fused pipeline for MI455X (gfx1250, wave32, WMMA, async-LDS)
//   B=4, T=8192, D=1024, C=8
//   Phase 0: one-time f32 -> bf16 conversion of x, W_in, W_gate (workspace)
//   Phase 1: dual GEMM (x@W_in^T, x@W_gate^T):
//            A tile (64x32 bf16) staged in LDS via async global->LDS copies,
//            double buffered; fragments via ds_load_b128;
//            8x v_wmma_f32_16x16x32_bf16 per K-step per wave;
//            fused epilogue -> alpha, xbeta (f32, workspace)
//   Phase 2: chunk-local scan aggregates (chunk=256, 32 chunks)
//   Phase 3: carry scan over chunk aggregates
//   Phase 4: seeded re-scan, write h
// ---------------------------------------------------------------------------

typedef __attribute__((ext_vector_type(16))) __bf16 v16bf;
typedef __attribute__((ext_vector_type(8)))  __bf16 bf16x8;
typedef __attribute__((ext_vector_type(8)))  float  v8f;
typedef __attribute__((ext_vector_type(8)))  float  f32x8;

#define DCONST   1024
#define TCONST   8192
#define BCONST   4
#define BT       32768          // B*T
#define CHUNK    256
#define NCHUNK   32             // T / CHUNK
#define AROWS    64             // A tile rows per block
#define APAD     40             // padded row stride (bf16) for bank spread

#if __has_builtin(__builtin_amdgcn_global_load_async_to_lds_b128)
#define ASYNC_LDS 1
// Builtin signature (from hipcc diagnostic): param0 = int4 addrspace(1)*,
// then LDS dst, imm offset, imm cpol.
typedef int async_b128_t __attribute__((vector_size(16)));
typedef async_b128_t __attribute__((address_space(1)))* async_gptr_t;
typedef async_b128_t __attribute__((address_space(3)))* async_lptr_t;
#else
#define ASYNC_LDS 0
#endif

// 16-byte global -> LDS copy (async when available; sync fallback).
static __device__ __forceinline__ void copy16_g2l(const __bf16* g, __bf16* l) {
#if ASYNC_LDS
    __builtin_amdgcn_global_load_async_to_lds_b128(
        (async_gptr_t)(unsigned long long)g,
        (async_lptr_t)(unsigned)(unsigned long long)l,
        0, 0);
#else
    *(bf16x8*)l = *(const bf16x8*)g;   // global_load_b128 + ds_store_b128
#endif
}

static __device__ __forceinline__ void wait_async_lds() {
#if ASYNC_LDS
#if __has_builtin(__builtin_amdgcn_s_wait_asynccnt)
    __builtin_amdgcn_s_wait_asynccnt(0);
#else
    asm volatile("s_wait_asynccnt 0x0" ::: "memory");
#endif
#endif
}

// ---------------------------------------------------------------------------
// Phase 0: f32 -> bf16 bulk conversion (packed, 8 elements/thread).
// ---------------------------------------------------------------------------
__global__ __launch_bounds__(256) void cvt_f32_to_bf16(
    const float* __restrict__ in, __bf16* __restrict__ out, unsigned n)
{
    const size_t i = ((size_t)blockIdx.x * blockDim.x + threadIdx.x) * 8;
    if (i + 8 > n) return;
    float4 a = *(const float4*)(in + i);
    float4 b = *(const float4*)(in + i + 4);
    f32x8 v;
    v[0] = a.x; v[1] = a.y; v[2] = a.z; v[3] = a.w;
    v[4] = b.x; v[5] = b.y; v[6] = b.z; v[7] = b.w;
    bf16x8 o = __builtin_convertvector(v, bf16x8);
    *(bf16x8*)(out + i) = o;
}

// Load one 16-element bf16 fragment per lane following the ISA layout:
// elements 0..7  -> K = kbase+klo .. +7        (VGPRs 0..3)
// elements 8..15 -> K = kbase+16+klo .. +7     (VGPRs 4..7)
// p points at (row base + kbase + klo); reads p[0..7] and p[16..23].
// Works for both global (global_load_b128) and LDS (ds_load_b128) pointers.
static __device__ __forceinline__ v16bf load_frag(const __bf16* p) {
    bf16x8 lo = *(const bf16x8*)(p);
    bf16x8 hi = *(const bf16x8*)(p + 16);
    return __builtin_shufflevector(lo, hi,
                                   0, 1, 2, 3, 4, 5, 6, 7,
                                   8, 9, 10, 11, 12, 13, 14, 15);
}

static __device__ __forceinline__ float sigmoidf_(float v) {
    return 1.0f / (1.0f + __expf(-v));
}

// ---------------------------------------------------------------------------
// Phase 1: dual-GEMM + elementwise epilogue.
// Block: 256 threads = 8 waves. Block tile: M=64 rows x N=128 cols.
// A tile staged in LDS (async, double buffered); B streamed from global.
// Wave w: N-tile [Nbase, Nbase+16), four M-subtiles, two weight matrices
//   -> 8 f32 accumulators (64 VGPRs), 8 WMMAs per K=32 step.
// ---------------------------------------------------------------------------
__global__ __launch_bounds__(256) void rg_lru_gates(
    const float*  __restrict__ x,        // [BT, D] f32 (epilogue only)
    const __bf16* __restrict__ xb,       // [BT, D] bf16
    const __bf16* __restrict__ Wi,       // [D, D] bf16 row-major (row = out dim)
    const __bf16* __restrict__ Wg,       // [D, D] bf16
    const float*  __restrict__ b_in,     // [D]
    const float*  __restrict__ b_gate,   // [D]
    const float*  __restrict__ lam,      // [D]
    float* __restrict__ wsA,             // [BT, D] alpha out
    float* __restrict__ wsXB)            // [BT, D] xbeta out
{
    __shared__ __align__(16) __bf16 smA[2][AROWS * APAD];   // 2 x 5 KB

    const int tid  = threadIdx.x;
    const int lane = tid & 31;
    const int wave = tid >> 5;
    const int half = lane >> 4;         // 0: lanes 0-15, 1: lanes 16-31
    const int lrow = lane & 15;
    const int klo  = half * 8;          // per-lane K sub-offset

    const int Mbase = blockIdx.y * AROWS;
    const int Nbase = blockIdx.x * 128 + wave * 16;

    // Cooperative A-tile copy mapping: 256 threads x 16B = 64 rows x 32 bf16.
    const int crow = tid >> 2;           // 0..63
    const int cseg = (tid & 3) * 8;      // bf16 offset within row: 0,8,16,24
    const __bf16* gA = xb + (size_t)(Mbase + crow) * DCONST + cseg;
    __bf16* lA0 = &smA[0][crow * APAD + cseg];
    __bf16* lA1 = &smA[1][crow * APAD + cseg];

    const __bf16* pbi = Wi + (size_t)(Nbase + lrow) * DCONST + klo;
    const __bf16* pbg = Wg + (size_t)(Nbase + lrow) * DCONST + klo;

    v8f acc_in[4] = {v8f{}, v8f{}, v8f{}, v8f{}};
    v8f acc_rg[4] = {v8f{}, v8f{}, v8f{}, v8f{}};

    // Preload tile 0.
    copy16_g2l(gA, lA0);

    for (int kk = 0; kk < DCONST / 32; ++kk) {
        wait_async_lds();
        __syncthreads();                 // tile kk resident in smA[kk&1]

        if (kk + 1 < DCONST / 32)        // prefetch next tile into other buffer
            copy16_g2l(gA + (kk + 1) * 32, (kk & 1) ? lA0 : lA1);

        const int k0 = kk * 32;
        __builtin_prefetch(pbi + k0 + 64, 0, 3);   // weight stream lookahead
        __builtin_prefetch(pbg + k0 + 64, 0, 3);

        v16bf bi = load_frag(pbi + k0);
        v16bf bg = load_frag(pbg + k0);
        const __bf16* Ab = smA[kk & 1];
        #pragma unroll
        for (int mt = 0; mt < 4; ++mt) {
            v16bf a = load_frag(Ab + (mt * 16 + lrow) * APAD + klo);  // ds_load_b128
            acc_in[mt] = __builtin_amdgcn_wmma_f32_16x16x32_bf16(
                false, a, false, bi, (short)0, acc_in[mt], false, false);
            acc_rg[mt] = __builtin_amdgcn_wmma_f32_16x16x32_bf16(
                false, a, false, bg, (short)0, acc_rg[mt], false, false);
        }
        __syncthreads();                 // readers done before buffer reuse
    }

    // Epilogue. D-matrix layout: lane -> n = lane&15; VGPR j -> m = j + 8*half.
    const int d = Nbase + lrow;
    const float bi_d   = b_in[d];
    const float bg_d   = b_gate[d];
    const float lam_d  = lam[d];
    const float sp     = log1pf(__expf(lam_d));   // softplus(forget_lambda)
    const float negCsp = -8.0f * sp;              // -C * softplus

    #pragma unroll
    for (int mt = 0; mt < 4; ++mt) {
        #pragma unroll
        for (int j = 0; j < 8; ++j) {
            const int row = Mbase + mt * 16 + half * 8 + j;
            const size_t idx = (size_t)row * DCONST + d;
            const float ig = acc_in[mt][j] + bi_d;
            const float rg = acc_rg[mt][j] + bg_d;
            const float alpha = __expf(negCsp * sigmoidf_(rg));
            const float beta  = sqrtf(1.0f - alpha * alpha + 1e-6f);
            const float xv    = x[idx];
            wsA[idx]  = alpha;
            wsXB[idx] = beta * sigmoidf_(ig) * xv;
        }
    }
}

// ---------------------------------------------------------------------------
// Phase 2: per-(b, chunk, d) local scan -> (prod alpha, local h at chunk end).
// Lanes map to consecutive d -> fully coalesced 1KB rows per step.
// ---------------------------------------------------------------------------
__global__ __launch_bounds__(256) void rg_lru_chunk_reduce(
    const float* __restrict__ wsA, const float* __restrict__ wsXB,
    float* __restrict__ aggA, float* __restrict__ aggB)
{
    const int d = blockIdx.x * blockDim.x + threadIdx.x;   // 0..1023
    const int c = blockIdx.y;                              // 0..31
    const int b = blockIdx.z;                              // 0..3
    size_t base = ((size_t)b * TCONST + (size_t)c * CHUNK) * DCONST + d;
    float aprod = 1.0f, h = 0.0f;
    for (int i = 0; i < CHUNK; ++i) {
        const float a   = wsA[base];
        const float xbv = wsXB[base];
        h = fmaf(a, h, xbv);
        aprod *= a;
        base += DCONST;
    }
    const size_t agg = ((size_t)b * NCHUNK + c) * DCONST + d;
    aggA[agg] = aprod;
    aggB[agg] = h;
}

// ---------------------------------------------------------------------------
// Phase 3: exclusive scan over the 32 chunk aggregates per (b, d) -> carries.
// ---------------------------------------------------------------------------
__global__ __launch_bounds__(256) void rg_lru_carry_scan(
    const float* __restrict__ aggA, const float* __restrict__ aggB,
    float* __restrict__ carry)
{
    const int d = blockIdx.x * blockDim.x + threadIdx.x;   // 0..1023
    const int b = blockIdx.y;                              // 0..3
    float h = 0.0f;
    for (int c = 0; c < NCHUNK; ++c) {
        const size_t idx = ((size_t)b * NCHUNK + c) * DCONST + d;
        carry[idx] = h;                       // h entering this chunk
        h = fmaf(aggA[idx], h, aggB[idx]);
    }
}

// ---------------------------------------------------------------------------
// Phase 4: seeded re-scan of each chunk, write h.
// ---------------------------------------------------------------------------
__global__ __launch_bounds__(256) void rg_lru_final(
    const float* __restrict__ wsA, const float* __restrict__ wsXB,
    const float* __restrict__ carry, float* __restrict__ out)
{
    const int d = blockIdx.x * blockDim.x + threadIdx.x;
    const int c = blockIdx.y;
    const int b = blockIdx.z;
    float h = carry[((size_t)b * NCHUNK + c) * DCONST + d];
    size_t base = ((size_t)b * TCONST + (size_t)c * CHUNK) * DCONST + d;
    for (int i = 0; i < CHUNK; ++i) {
        h = fmaf(wsA[base], h, wsXB[base]);
        out[base] = h;
        base += DCONST;
    }
}

// ---------------------------------------------------------------------------
extern "C" void kernel_launch(void* const* d_in, const int* in_sizes, int n_in,
                              void* d_out, int out_size, void* d_ws, size_t ws_size,
                              hipStream_t stream) {
    (void)in_sizes; (void)n_in; (void)out_size; (void)ws_size;
    const float* x      = (const float*)d_in[0];
    const float* W_in   = (const float*)d_in[1];
    const float* b_in   = (const float*)d_in[2];
    const float* W_gate = (const float*)d_in[3];
    const float* b_gate = (const float*)d_in[4];
    const float* lam    = (const float*)d_in[5];
    float* out = (float*)d_out;

    const size_t NTD = (size_t)BT * DCONST;                // 33,554,432
    const size_t NAG = (size_t)BCONST * NCHUNK * DCONST;   // 131,072
    const size_t NW  = (size_t)DCONST * DCONST;            // 1,048,576
    float* ws    = (float*)d_ws;
    float* wsA   = ws;                   // alpha  [BT, D]
    float* wsXB  = ws + NTD;             // xbeta  [BT, D]
    float* aggA  = wsXB + NTD;           // chunk prod(alpha) [B, 32, D]
    float* aggB  = aggA + NAG;           // chunk local h     [B, 32, D]
    float* carry = aggB + NAG;           // carries           [B, 32, D]
    __bf16* x_bf  = (__bf16*)(carry + NAG);  // [BT, D] bf16
    __bf16* Wi_bf = x_bf + NTD;              // [D, D] bf16
    __bf16* Wg_bf = Wi_bf + NW;              // [D, D] bf16

    // Phase 0: bf16 conversions (one-time).
    cvt_f32_to_bf16<<<(unsigned)(NTD / 8 / 256), 256, 0, stream>>>(x, x_bf, (unsigned)NTD);
    cvt_f32_to_bf16<<<(unsigned)(NW / 8 / 256), 256, 0, stream>>>(W_in, Wi_bf, (unsigned)NW);
    cvt_f32_to_bf16<<<(unsigned)(NW / 8 / 256), 256, 0, stream>>>(W_gate, Wg_bf, (unsigned)NW);

    // Phase 1: dual-GEMM + epilogue. Grid: N-blocks x M-blocks.
    rg_lru_gates<<<dim3(DCONST / 128, BT / AROWS), 256, 0, stream>>>(
        x, x_bf, Wi_bf, Wg_bf, b_in, b_gate, lam, wsA, wsXB);

    // Phase 2: chunk aggregates.
    rg_lru_chunk_reduce<<<dim3(DCONST / 256, NCHUNK, BCONST), 256, 0, stream>>>(
        wsA, wsXB, aggA, aggB);

    // Phase 3: carry scan.
    rg_lru_carry_scan<<<dim3(DCONST / 256, BCONST), 256, 0, stream>>>(
        aggA, aggB, carry);

    // Phase 4: seeded scan, write h.
    rg_lru_final<<<dim3(DCONST / 256, NCHUNK, BCONST), 256, 0, stream>>>(
        wsA, wsXB, carry, out);
}
